// DecoderBlock_26113401160420
// MI455X (gfx1250) — matrix operand order repaired and checked by reading, not compiled, not run on hardware
//
#include <hip/hip_runtime.h>

// ---------------- problem constants (fixed by reference) ----------------
#define B_    2
#define S_    2048
#define D_    2048
#define H_    16
#define DH_   128
#define DFF_  8192
#define BS_   (B_ * S_)          // 4096 rows
#define EPS_  1e-5f
#define SCALE_  0.022097086912079608f  // 1/sqrt(2048) (reference scales by sqrt(D))
#define SCALE2_ 0.031879357673f        // log2(e)/sqrt(2048): fold exp's log2e mul
#define CH_   128                 // scan chunk length; S_/CH_ = 16
#define NCH_  (S_ / CH_)

// ---------------- vector types ----------------
typedef __attribute__((ext_vector_type(16))) __bf16 v16bf;
typedef __attribute__((ext_vector_type(8)))  __bf16 v8bf;
typedef __attribute__((ext_vector_type(4)))  __bf16 v4bf;
typedef __attribute__((ext_vector_type(8)))  float  v8f;
typedef __attribute__((ext_vector_type(4)))  float  f4;

// native conversions -> v_cvt_*bf16* instead of scalar bit math
__device__ __forceinline__ __bf16 f2bf(float f) { return (__bf16)f; }
__device__ __forceinline__ float bf2f(__bf16 b) { return (float)b; }

// raw v_exp_f32 (exp2), no libm denormal-range fixup; args here are O(1)
__device__ __forceinline__ float fast_exp2(float x) {
#if __has_builtin(__builtin_amdgcn_exp2f)
  return __builtin_amdgcn_exp2f(x);
#else
  return exp2f(x);
#endif
}

__device__ __forceinline__ v8f v8f_zero() {
  v8f z;
#pragma unroll
  for (int i = 0; i < 8; ++i) z[i] = 0.f;
  return z;
}

// A-fragment (16x32 bf16, M x K): lane l holds row (l&15); K chunks at
// kb=(l>>4)*8 (elements 0..7) and kb+16 (elements 8..15).  [ISA 7.12.2]
__device__ __forceinline__ v16bf load_frag_a(const __bf16* row_ptr, int lane) {
  int kb = (lane >> 4) * 8;
  v8bf lo = *(const v8bf*)(row_ptr + kb);
  v8bf hi = *(const v8bf*)(row_ptr + kb + 16);
  return __builtin_shufflevector(lo, hi, 0,1,2,3,4,5,6,7,8,9,10,11,12,13,14,15);
}
// B-fragment (32x16 bf16, K x N) from N-major [n][k] storage:
// lane l holds column (l&15); K chunks at kc=(l>>4)*8 (elements 0..7) and kc+16 (elements 8..15).
__device__ __forceinline__ v16bf load_frag_b(const __bf16* row_ptr, int lane) {
  int kc = (lane >> 4) * 8;
  v8bf lo = *(const v8bf*)(row_ptr + kc);
  v8bf hi = *(const v8bf*)(row_ptr + kc + 16);
  return __builtin_shufflevector(lo, hi, 0,1,2,3,4,5,6,7,8,9,10,11,12,13,14,15);
}

// ---------------- weight convert + transpose: W[k][n] f32 -> Wt[n][k] bf16 ----
__global__ __launch_bounds__(256) void wconv_kernel(
    const float* __restrict__ W, __bf16* __restrict__ Wt, int K, int N) {
  __shared__ __bf16 sT[64][72];
  int k0 = blockIdx.y * 64, n0 = blockIdx.x * 64;
  int t = threadIdx.x;
  int kk = t >> 2, nn = (t & 3) * 16;
  const float* wp = W + (size_t)(k0 + kk) * N + n0 + nn;
#pragma unroll
  for (int i = 0; i < 16; i += 4) {
    f4 v = *(const f4*)(wp + i);
    v4bf c = __builtin_convertvector(v, v4bf);
#pragma unroll
    for (int j = 0; j < 4; ++j) sT[nn + i + j][kk] = c[j];
  }
  __syncthreads();
  int nn2 = t >> 2, kc = (t & 3) * 16;
  __bf16* op = Wt + (size_t)(n0 + nn2) * K + k0 + kc;
  *(v8bf*)op       = *(const v8bf*)&sT[nn2][kc];
  *(v8bf*)(op + 8) = *(const v8bf*)&sT[nn2][kc + 8];
}

// ---------------- LayerNorm: f32 row -> bf16 row ----------------
__global__ __launch_bounds__(256) void ln_kernel(
    const float* __restrict__ x, const float* __restrict__ g,
    const float* __restrict__ b, __bf16* __restrict__ out) {
  __shared__ float sbuf[16];
  int row = blockIdx.x;
  const float* xr = x + (size_t)row * D_;
  float v[8];
  float s = 0.f;
#pragma unroll
  for (int i = 0; i < 8; ++i) { v[i] = xr[threadIdx.x + i * 256]; s += v[i]; }
#pragma unroll
  for (int m = 16; m >= 1; m >>= 1) s += __shfl_xor(s, m, 32);
  int w = threadIdx.x >> 5, lane = threadIdx.x & 31;
  if (lane == 0) sbuf[w] = s;
  __syncthreads();
  if (threadIdx.x == 0) { float t = 0; for (int i = 0; i < 8; ++i) t += sbuf[i]; sbuf[8] = t; }
  __syncthreads();
  float mu = sbuf[8] * (1.0f / D_);
  float s2 = 0.f;
#pragma unroll
  for (int i = 0; i < 8; ++i) { float d = v[i] - mu; s2 += d * d; }
#pragma unroll
  for (int m = 16; m >= 1; m >>= 1) s2 += __shfl_xor(s2, m, 32);
  __syncthreads();
  if (lane == 0) sbuf[w] = s2;
  __syncthreads();
  if (threadIdx.x == 0) { float t = 0; for (int i = 0; i < 8; ++i) t += sbuf[i]; sbuf[8] = t; }
  __syncthreads();
  float rr = rsqrtf(sbuf[8] * (1.0f / D_) + EPS_);
#pragma unroll
  for (int i = 0; i < 8; ++i) {
    int c = threadIdx.x + i * 256;
    out[(size_t)row * D_ + c] = f2bf((v[i] - mu) * rr * g[c] + b[c]);
  }
}

// ---------------- bf16 WMMA GEMM, double-buffered LDS, 1 barrier/K-step -----
#define BM 128
#define BN 128
#define BK 32
#define LP 40   // padded LDS stride (elements)

// mode: 0 = f32 out [M,N], 1 = bf16 out [M,N], 2 = bf16 out scattered to [B,H,S,DH]
template <bool USE_WT>
__global__ __launch_bounds__(256) void gemm_bf16(
    const __bf16* __restrict__ A, const __bf16* __restrict__ Wt,
    const float* __restrict__ Wf, const float* __restrict__ bias,
    const float* __restrict__ resid, float* __restrict__ outf,
    __bf16* __restrict__ outb, int M, int N, int K, int mode, int relu) {
  __shared__ __bf16 sA[2][BM * LP];
  __shared__ __bf16 sB[2][BN * LP];
  int tid = threadIdx.x;
  int m0 = blockIdx.y * BM, n0 = blockIdx.x * BN;
  int w = tid >> 5, lane = tid & 31;
  int wM = (w & 3) * 32, wN = (w >> 2) * 64;
  int lr = lane & 15;

  v8f acc[2][4];
#pragma unroll
  for (int i = 0; i < 2; ++i)
#pragma unroll
    for (int j = 0; j < 4; ++j) acc[i][j] = v8f_zero();

  int ar = tid >> 1, ac = (tid & 1) * 16;  // 128 rows x 32 cols tile mapping
  const __bf16* aptr = A + (size_t)(m0 + ar) * K + ac;
  const __bf16* btptr = Wt + (size_t)(n0 + ar) * K + ac;     // [n][k] bf16
  int fk = tid & 31, fn = (tid >> 5) * 16;                   // fallback f32 path
  const float* fptr = Wf + (size_t)fk * N + n0 + fn;

  v8bf a0, a1, b0, b1;
  f4 fw[4];

  auto fetch = [&](int k0) {
    a0 = *(const v8bf*)(aptr + k0);
    a1 = *(const v8bf*)(aptr + k0 + 8);
    if (USE_WT) {
      b0 = *(const v8bf*)(btptr + k0);
      b1 = *(const v8bf*)(btptr + k0 + 8);
    } else {
      const float* p = fptr + (size_t)k0 * N;
#pragma unroll
      for (int i = 0; i < 4; ++i) fw[i] = *(const f4*)(p + i * 4);
    }
  };
  auto stage = [&](int buf) {
    *(v8bf*)(&sA[buf][ar * LP + ac])     = a0;
    *(v8bf*)(&sA[buf][ar * LP + ac + 8]) = a1;
    if (USE_WT) {
      *(v8bf*)(&sB[buf][ar * LP + ac])     = b0;
      *(v8bf*)(&sB[buf][ar * LP + ac + 8]) = b1;
    } else {
#pragma unroll
      for (int i = 0; i < 4; ++i) {
        v4bf c = __builtin_convertvector(fw[i], v4bf);
#pragma unroll
        for (int j = 0; j < 4; ++j) sB[buf][(fn + i * 4 + j) * LP + fk] = c[j];
      }
    }
  };

  fetch(0);
  stage(0);
  int buf = 0;
  for (int k0 = 0;;) {
    __syncthreads();               // staged tile visible; prior reads done
    bool has_next = (k0 + BK) < K;
    if (has_next) fetch(k0 + BK);  // global loads overlap WMMA below
    v16bf af[2], bfr[4];
#pragma unroll
    for (int ti = 0; ti < 2; ++ti)
      af[ti] = load_frag_a(&sA[buf][(wM + ti * 16 + lr) * LP], lane);
#pragma unroll
    for (int tj = 0; tj < 4; ++tj)
      bfr[tj] = load_frag_b(&sB[buf][(wN + tj * 16 + lr) * LP], lane);
#pragma unroll
    for (int ti = 0; ti < 2; ++ti)
#pragma unroll
      for (int tj = 0; tj < 4; ++tj)
        acc[ti][tj] = __builtin_amdgcn_wmma_f32_16x16x32_bf16(
            false, af[ti], false, bfr[tj], (short)0, acc[ti][tj], false, false);
    if (!has_next) break;
    stage(buf ^ 1);
    buf ^= 1;
    k0 += BK;
  }

  int rhalf = (lane >> 4) * 8;
#pragma unroll
  for (int ti = 0; ti < 2; ++ti)
#pragma unroll
    for (int tj = 0; tj < 4; ++tj)
#pragma unroll
      for (int j = 0; j < 8; ++j) {
        int m = m0 + wM + ti * 16 + rhalf + j;
        int n = n0 + wN + tj * 16 + lr;
        float val = acc[ti][tj][j] + bias[n];
        if (relu) val = fmaxf(val, 0.f);
        if (resid) val += resid[(size_t)m * N + n];
        if (mode == 0) {
          outf[(size_t)m * N + n] = val;
        } else if (mode == 1) {
          outb[(size_t)m * N + n] = f2bf(val);
        } else {  // scatter [BS, D] -> [B, H, S, DH]
          int bb = m >> 11, ss = m & (S_ - 1);
          int hh = n >> 7,  dd = n & (DH_ - 1);
          outb[((size_t)(bb * H_ + hh) * S_ + ss) * DH_ + dd] = f2bf(val);
        }
      }
}

// ---------------- chunked exclusive prefix-sum of V over sequence ------------
__global__ void vscan_chunk(const __bf16* __restrict__ v, float* __restrict__ chunksum) {
  int bc = blockIdx.x;                   // bh*NCH_ + chunk
  int d = threadIdx.x;                   // 128 = DH_
  size_t base = (size_t)bc * CH_ * DH_ + d;
  float s = 0.f;
  for (int i = 0; i < CH_; ++i) s += bf2f(v[base + (size_t)i * DH_]);
  chunksum[(size_t)bc * DH_ + d] = s;
}
__global__ void vscan_base(float* __restrict__ chunksum) {  // in-place exclusive scan over chunks
  int bh = blockIdx.x;
  int d = threadIdx.x;
  float acc = 0.f;
  for (int c = 0; c < NCH_; ++c) {
    size_t i = ((size_t)bh * NCH_ + c) * DH_ + d;
    float t = chunksum[i];
    chunksum[i] = acc;
    acc += t;
  }
}
__global__ void vscan_prefix(const __bf16* __restrict__ v,
                             const float* __restrict__ chunkbase,
                             float* __restrict__ prefix) {
  int bc = blockIdx.x;
  int d = threadIdx.x;
  float acc = chunkbase[(size_t)bc * DH_ + d];
  size_t base = (size_t)bc * CH_ * DH_ + d;
  for (int i = 0; i < CH_; ++i) {
    size_t idx = base + (size_t)i * DH_;
    prefix[idx] = acc;
    acc += bf2f(v[idx]);
  }
}

// ---------------- diagonal scores: diag[r] = exp(q_r . k_r / sqrt(D)) --------
__global__ __launch_bounds__(256) void attn_diag_kernel(
    const __bf16* __restrict__ q, const __bf16* __restrict__ k,
    float* __restrict__ diag) {
  int r = (int)((blockIdx.x * 256 + threadIdx.x) >> 5);  // row = bh*S + s
  int lane = threadIdx.x & 31;
  v4bf qv = *(const v4bf*)(q + (size_t)r * DH_ + lane * 4);
  v4bf kv = *(const v4bf*)(k + (size_t)r * DH_ + lane * 4);
  float s = 0.f;
#pragma unroll
  for (int i = 0; i < 4; ++i) s += bf2f(qv[i]) * bf2f(kv[i]);
#pragma unroll
  for (int m = 16; m >= 1; m >>= 1) s += __shfl_xor(s, m, 32);
  if (lane == 0) diag[r] = fast_exp2(s * SCALE2_);
}

// ---------------- upper-triangular exp row-sums via WMMA (branchless) --------
// denom[q] = q + sum_{a>=q} exp(q_q . k_a / sqrt(D))
__global__ __launch_bounds__(256) void attn_rowsum_kernel(
    const __bf16* __restrict__ q, const __bf16* __restrict__ k,
    float* __restrict__ denom) {
  int bh = blockIdx.y;
  int w = threadIdx.x >> 5, lane = threadIdx.x & 31;
  int lr = lane & 15;
  int rh = (lane >> 4) * 8;
  int qb = blockIdx.x * 128 + w * 16;

  const __bf16* qrow = q + ((size_t)bh * S_ + qb + lr) * DH_;
  v16bf qf[4];
#pragma unroll
  for (int kk = 0; kk < 4; ++kk) qf[kk] = load_frag_a(qrow + kk * 32, lane);

  float rows[8];
#pragma unroll
  for (int j = 0; j < 8; ++j) rows[j] = 0.f;

  const __bf16* kbase = k + (size_t)bh * S_ * DH_;

  auto score_tile = [&](int a0) -> v8f {
    const __bf16* krow = kbase + (size_t)(a0 + lr) * DH_;
    v16bf bfr[4];
#pragma unroll
    for (int kk = 0; kk < 4; ++kk) bfr[kk] = load_frag_b(krow + kk * 32, lane);
    v8f acc = v8f_zero();
#pragma unroll
    for (int kk = 0; kk < 4; ++kk)
      acc = __builtin_amdgcn_wmma_f32_16x16x32_bf16(
          false, qf[kk], false, bfr[kk], (short)0, acc, false, false);
    return acc;
  };

  {  // diagonal tile: mask a >= q, branchless select
    v8f acc = score_tile(qb);
    int a = qb + lr;
#pragma unroll
    for (int j = 0; j < 8; ++j) {
      float e = fast_exp2(acc[j] * SCALE2_);   // raw v_exp_f32
      rows[j] += (a >= qb + rh + j) ? e : 0.f;
    }
  }
  for (int a0 = qb + 16; a0 < S_; a0 += 16) {  // strictly above diagonal: no mask
    v8f acc = score_tile(a0);
#pragma unroll
    for (int j = 0; j < 8; ++j) rows[j] += fast_exp2(acc[j] * SCALE2_);
  }

#pragma unroll
  for (int j = 0; j < 8; ++j) {
    float r = rows[j];
    r += __shfl_xor(r, 1, 16); r += __shfl_xor(r, 2, 16);
    r += __shfl_xor(r, 4, 16); r += __shfl_xor(r, 8, 16);
    if (lr == 0) {
      int qi = qb + rh + j;
      denom[(size_t)bh * S_ + qi] = (float)qi + r;
    }
  }
}

// ---------------- attn = (prefixV + diag*v) / denom, scatter to [B,S,D] bf16 --
__global__ void attn_combine_kernel(
    const __bf16* __restrict__ v, const float* __restrict__ prefix,
    const float* __restrict__ denom, const float* __restrict__ diag,
    __bf16* __restrict__ attnbf) {
  size_t idx = (size_t)blockIdx.x * blockDim.x + threadIdx.x;  // over B*H*S*DH
  int d = (int)(idx & (DH_ - 1));
  int s = (int)((idx >> 7) & (S_ - 1));
  int bh = (int)(idx >> 18);               // S_*DH_ = 2^18
  float num = prefix[idx] + diag[(size_t)bh * S_ + s] * bf2f(v[idx]);
  float val = num / denom[(size_t)bh * S_ + s];
  int b = bh >> 4, h = bh & (H_ - 1);
  attnbf[((size_t)(b * S_ + s)) * D_ + h * DH_ + d] = f2bf(val);
}

// ---------------- driver ----------------
extern "C" void kernel_launch(void* const* d_in, const int* in_sizes, int n_in,
                              void* d_out, int out_size, void* d_ws, size_t ws_size,
                              hipStream_t stream) {
  (void)in_sizes; (void)n_in; (void)out_size;
  const float* x    = (const float*)d_in[0];
  const float* wq   = (const float*)d_in[1];
  const float* bq   = (const float*)d_in[2];
  const float* wk   = (const float*)d_in[3];
  const float* bk   = (const float*)d_in[4];
  const float* wv   = (const float*)d_in[5];
  const float* bv   = (const float*)d_in[6];
  const float* wo   = (const float*)d_in[7];
  const float* bo   = (const float*)d_in[8];
  const float* ln1g = (const float*)d_in[9];
  const float* ln1b = (const float*)d_in[10];
  const float* w1   = (const float*)d_in[11];
  const float* b1   = (const float*)d_in[12];
  const float* w2   = (const float*)d_in[13];
  const float* b2   = (const float*)d_in[14];
  const float* ln2g = (const float*)d_in[15];
  const float* ln2b = (const float*)d_in[16];
  float* outp = (float*)d_out;

  char* ws = (char*)d_ws;
  size_t off = 0;
  auto alloc = [&](size_t bytes) -> void* {
    void* p = ws + off;
    off = (off + bytes + 255) & ~(size_t)255;
    return p;
  };
  const size_t nBSD = (size_t)BS_ * D_;
  __bf16* hbf    = (__bf16*)alloc(nBSD * 2);
  __bf16* qbuf   = (__bf16*)alloc(nBSD * 2);
  __bf16* kbuf   = (__bf16*)alloc(nBSD * 2);
  __bf16* vbuf   = (__bf16*)alloc(nBSD * 2);
  __bf16* attnbf = (__bf16*)alloc(nBSD * 2);
  __bf16* ffmid  = (__bf16*)alloc((size_t)BS_ * DFF_ * 2);
  float*  prefix = (float*)alloc(nBSD * 4);
  float*  denom  = (float*)alloc((size_t)B_ * H_ * S_ * 4);
  float*  diag   = (float*)alloc((size_t)B_ * H_ * S_ * 4);
  float*  csum   = (float*)alloc((size_t)B_ * H_ * NCH_ * DH_ * 4);

  // optional pre-converted [N][K] bf16 weights (gated on ws_size; deterministic)
  size_t wt_need = 4 * ((size_t)D_ * D_ * 2 + 256) + 2 * ((size_t)D_ * DFF_ * 2 + 256);
  bool use_wt = (off + wt_need) <= ws_size;
  __bf16 *wqT = nullptr, *wkT = nullptr, *wvT = nullptr, *woT = nullptr,
         *w1T = nullptr, *w2T = nullptr;
  if (use_wt) {
    wqT = (__bf16*)alloc((size_t)D_ * D_ * 2);
    wkT = (__bf16*)alloc((size_t)D_ * D_ * 2);
    wvT = (__bf16*)alloc((size_t)D_ * D_ * 2);
    woT = (__bf16*)alloc((size_t)D_ * D_ * 2);
    w1T = (__bf16*)alloc((size_t)D_ * DFF_ * 2);
    w2T = (__bf16*)alloc((size_t)DFF_ * D_ * 2);
    dim3 gDD(D_ / 64, D_ / 64);
    wconv_kernel<<<gDD, 256, 0, stream>>>(wq, wqT, D_, D_);
    wconv_kernel<<<gDD, 256, 0, stream>>>(wk, wkT, D_, D_);
    wconv_kernel<<<gDD, 256, 0, stream>>>(wv, wvT, D_, D_);
    wconv_kernel<<<gDD, 256, 0, stream>>>(wo, woT, D_, D_);
    wconv_kernel<<<dim3(DFF_ / 64, D_ / 64), 256, 0, stream>>>(w1, w1T, D_, DFF_);
    wconv_kernel<<<dim3(D_ / 64, DFF_ / 64), 256, 0, stream>>>(w2, w2T, DFF_, D_);
  }

  dim3 gD(D_ / BN, BS_ / BM);      // (16, 32)
  dim3 gF(DFF_ / BN, BS_ / BM);    // (64, 32)

  ln_kernel<<<BS_, 256, 0, stream>>>(x, ln1g, ln1b, hbf);
  if (use_wt) {
    gemm_bf16<true><<<gD, 256, 0, stream>>>(hbf, wqT, wq, bq, nullptr, nullptr, qbuf, BS_, D_, D_, 2, 0);
    gemm_bf16<true><<<gD, 256, 0, stream>>>(hbf, wkT, wk, bk, nullptr, nullptr, kbuf, BS_, D_, D_, 2, 0);
    gemm_bf16<true><<<gD, 256, 0, stream>>>(hbf, wvT, wv, bv, nullptr, nullptr, vbuf, BS_, D_, D_, 2, 0);
  } else {
    gemm_bf16<false><<<gD, 256, 0, stream>>>(hbf, nullptr, wq, bq, nullptr, nullptr, qbuf, BS_, D_, D_, 2, 0);
    gemm_bf16<false><<<gD, 256, 0, stream>>>(hbf, nullptr, wk, bk, nullptr, nullptr, kbuf, BS_, D_, D_, 2, 0);
    gemm_bf16<false><<<gD, 256, 0, stream>>>(hbf, nullptr, wv, bv, nullptr, nullptr, vbuf, BS_, D_, D_, 2, 0);
  }
  vscan_chunk<<<B_ * H_ * NCH_, DH_, 0, stream>>>(vbuf, csum);
  vscan_base<<<B_ * H_, DH_, 0, stream>>>(csum);
  vscan_prefix<<<B_ * H_ * NCH_, DH_, 0, stream>>>(vbuf, csum, prefix);
  attn_diag_kernel<<<(B_ * H_ * S_) / 8, 256, 0, stream>>>(qbuf, kbuf, diag);
  dim3 gR(S_ / 128, B_ * H_);
  attn_rowsum_kernel<<<gR, 256, 0, stream>>>(qbuf, kbuf, denom);
  attn_combine_kernel<<<(unsigned)(nBSD / 256), 256, 0, stream>>>(vbuf, prefix, denom, diag, attnbf);
  if (use_wt) {
    gemm_bf16<true><<<gD, 256, 0, stream>>>(attnbf, woT, wo, bo, x, outp, nullptr, BS_, D_, D_, 0, 0);
  } else {
    gemm_bf16<false><<<gD, 256, 0, stream>>>(attnbf, nullptr, wo, bo, x, outp, nullptr, BS_, D_, D_, 0, 0);
  }
  ln_kernel<<<BS_, 256, 0, stream>>>(outp, ln2g, ln2b, hbf);
  if (use_wt) {
    gemm_bf16<true><<<gF, 256, 0, stream>>>(hbf, w1T, w1, b1, nullptr, nullptr, ffmid, BS_, DFF_, D_, 1, 1);
    gemm_bf16<true><<<gD, 256, 0, stream>>>(ffmid, w2T, w2, b2, outp, outp, nullptr, BS_, D_, DFF_, 0, 0);
  } else {
    gemm_bf16<false><<<gF, 256, 0, stream>>>(hbf, nullptr, w1, b1, nullptr, nullptr, ffmid, BS_, DFF_, D_, 1, 1);
    gemm_bf16<false><<<gD, 256, 0, stream>>>(ffmid, nullptr, w2, b2, outp, outp, nullptr, BS_, D_, DFF_, 0, 0);
  }
}
